// ScannedLSTM_30975304139461
// MI455X (gfx1250) — matrix operand order repaired and checked by reading, not compiled, not run on hardware
//
#include <hip/hip_runtime.h>
#include <hip/hip_bf16.h>
#include <math.h>

// ---------------- problem constants (T=256, B=256, H=512) ----------------
#define T_  256
#define B_  256
#define H_  512
#define G4  (4 * H_)   // 2048 gate columns
#define K2  (2 * H_)   // 1024 = K of fused [x|h] @ [Wx;Wh]
#define NKC (K2 / 32)  // 32 K-chunks
#define MROWS 32       // batch rows staged per block (2 row-tiles)
#define AROW 1032      // padded LDS row stride (516 dwords -> 4-bank skew)

typedef __bf16 bf16_t;
typedef __attribute__((ext_vector_type(16))) __bf16 v16bf;
typedef __attribute__((ext_vector_type(8)))  __bf16 v8bf;
typedef __attribute__((ext_vector_type(8)))  float  v8f;

__device__ __forceinline__ float sigmoidf_(float x) {
    return 1.0f / (1.0f + __expf(-x));
}

// ---------------- fp32 -> bf16 streaming convert ----------------
__global__ __launch_bounds__(256)
void cvt_f32_to_bf16(const float* __restrict__ in, bf16_t* __restrict__ out, int n) {
    int i = blockIdx.x * blockDim.x + threadIdx.x;
    int stride = gridDim.x * blockDim.x;
    for (; i < n; i += stride) out[i] = (bf16_t)in[i];
}

// ---- pack [Wx;Wh] into WMMA-fragment-major bf16 layout:
//      Wfrag[kc][jTile][gate][lane][e], kc=K/32, jTile=j/16, lane=N%16+16*h, e=K slot (table order)
//      -> a B fragment is a contiguous, lane-coalesced 1KB block.
__global__ __launch_bounds__(256)
void pack_weights_frag(const float* __restrict__ Wx, const float* __restrict__ Wh,
                       bf16_t* __restrict__ Wfrag) {
    int i = blockIdx.x * blockDim.x + threadIdx.x;
    int stride = gridDim.x * blockDim.x;
    const int n = K2 * G4;                       // 2,097,152 elements
    for (; i < n; i += stride) {
        const int e    = i & 15;                 // K slot within lane
        const int lane = (i >> 4) & 31;          // N (lane&15), K half (lane>>4)
        const int gate = (i >> 9) & 3;
        const int jt   = (i >> 11) & 31;
        const int kc   = i >> 16;
        const int krow = kc * 32 + ((e < 8) ? (lane >> 4) * 8 + e : 16 + (lane >> 4) * 8 + (e - 8)); // 0..1023
        const int col  = gate * H_ + jt * 16 + (lane & 15);
        const float v  = (krow < H_) ? Wx[krow * G4 + col]
                                     : Wh[(krow - H_) * G4 + col];
        Wfrag[i] = (bf16_t)v;
    }
}

// ---------------- init c (fp32), h (bf16) state, and a zeroed dummy row ----------------
__global__ __launch_bounds__(256)
void init_state(const float* __restrict__ c0, const float* __restrict__ h0,
                float* __restrict__ c_state, bf16_t* __restrict__ h_bf,
                bf16_t* __restrict__ zero_row) {
    int i = blockIdx.x * blockDim.x + threadIdx.x;
    int stride = gridDim.x * blockDim.x;
    const int n = B_ * H_;
    for (int k = i; k < H_; k += stride) zero_row[k] = (bf16_t)0.0f;
    for (; i < n; i += stride) {
        c_state[i] = c0[i];
        h_bf[i]    = (bf16_t)h0[i];
    }
}

// ---------------- copy cT / hT into the output tail ----------------
__global__ __launch_bounds__(256)
void finalize_out(const float* __restrict__ c_state, const float* __restrict__ ys_last,
                  float* __restrict__ cT, float* __restrict__ hT) {
    int i = blockIdx.x * blockDim.x + threadIdx.x;
    int stride = gridDim.x * blockDim.x;
    const int n = B_ * H_;
    for (; i < n; i += stride) {
        cT[i] = c_state[i];
        hT[i] = ys_last[i];
    }
}

// ---- per-row-tile gate nonlinearity + state update
//      C/D layout: VGPR v, lane half hi -> row v+8*hi; N = lane&15.
__device__ __forceinline__ void gate_epilogue(
    v8f ai, v8f af, v8f ag, v8f ao,
    int rowBase, int hi, int jn,
    float bi, float bf, float bg, float bo,
    const int* __restrict__ resets_t, float* __restrict__ c_state,
    float* __restrict__ ys_t, bf16_t* __restrict__ h_next)
{
#pragma unroll
    for (int v = 0; v < 8; ++v) {
        const int r = rowBase + hi * 8 + v;
        const size_t idx = (size_t)r * H_ + jn;
        float c_old = c_state[idx];
        if (resets_t[r] != 0) c_old = 0.0f;
        const float zi = ai[v] + bi;
        const float zf = af[v] + bf;
        const float zg = ag[v] + bg;
        const float zo = ao[v] + bo;
        const float c_new = sigmoidf_(zf) * c_old + sigmoidf_(zi) * tanhf(zg);
        const float h_new = sigmoidf_(zo) * tanhf(c_new);
        c_state[idx] = c_new;
        ys_t[idx]    = h_new;
        h_next[idx]  = (bf16_t)h_new;
    }
}

// ---------------- one LSTM timestep: z = [x|h_prev] @ W + b ; gates ----------------
// Grid: 64 blocks x 128 threads (4 waves). Block stages 32 batch rows x 1024 K of
// [x|h] in LDS (66KB dynamic). Its 4 waves cover 4 hidden-col tiles; each wave
// accumulates 2 row-tiles x 4 gate tiles = 8 WMMAs per B-fetch (B reused 2x).
__global__ __launch_bounds__(128)
void lstm_step(const bf16_t* __restrict__ xt,        // B_ x H_   (ins_bf16 + t*B_*H_)
               const int*    __restrict__ resets_t,  // B_        (resets + t*B_)
               const bf16_t* __restrict__ Wfrag,     // fragment-major weights
               const float*  __restrict__ bias,      // G4
               const bf16_t* __restrict__ h_prev,    // B_ x H_
               const bf16_t* __restrict__ zero_row,  // H_ zeros (reset mask source)
               bf16_t*       __restrict__ h_next,    // B_ x H_
               float*        __restrict__ c_state,   // B_ x H_ (in-place)
               float*        __restrict__ ys_t)      // B_ x H_ (d_out + t*B_*H_)
{
    extern __shared__ __align__(16) bf16_t As[];      // MROWS * AROW bf16 (66KB)

    const int tid  = threadIdx.x;
    const int lane = tid & 31;
    const int wave = tid >> 5;                        // 0..3
    const int rowTile = blockIdx.x >> 3;              // 0..7 (32-row groups)
    const int colTile = (blockIdx.x & 7) * 4 + wave;  // 0..31
    const int row0 = rowTile * MROWS;
    const int j0   = colTile * 16;

    const int l  = lane & 15;                         // A row / D col within tile
    const int hi = lane >> 4;                         // lane half selects K sub-block

    // ---- cooperative LDS staging: 32 rows x (512 x | 512 h) bf16, coalesced v8bf
#pragma unroll
    for (int it = 0; it < 16; ++it) {
        const int v   = tid + it * 128;               // 0..2047
        const int row = v >> 6;                       // 0..31
        const int kv  = (v & 63) * 8;                 // 0..504
        *(v8bf*)&As[row * AROW + kv] =
            *(const v8bf*)(xt + (size_t)(row0 + row) * H_ + kv);
    }
#pragma unroll
    for (int it = 0; it < 16; ++it) {
        const int v   = tid + it * 128;
        const int row = v >> 6;
        const int kv  = (v & 63) * 8;
        const bf16_t* src = (resets_t[row0 + row] != 0)
                              ? (zero_row + kv)
                              : (h_prev + (size_t)(row0 + row) * H_ + kv);
        *(v8bf*)&As[row * AROW + H_ + kv] = *(const v8bf*)src;
    }
    __syncthreads();

    v8f acc_i0 = {}, acc_f0 = {}, acc_g0 = {}, acc_o0 = {};
    v8f acc_i1 = {}, acc_f1 = {}, acc_g1 = {}, acc_o1 = {};

    // B fragment base: chunk stride 65536 elems, gate stride 512 elems
    const bf16_t* bbase = Wfrag + (size_t)colTile * 2048 + (size_t)lane * 16;

    v16bf b0c = *(const v16bf*)(bbase);
    v16bf b1c = *(const v16bf*)(bbase + 512);
    v16bf b2c = *(const v16bf*)(bbase + 1024);
    v16bf b3c = *(const v16bf*)(bbase + 1536);

#pragma unroll 4
    for (int kc = 0; kc < NKC; ++kc) {
        // prefetch next chunk's B fragments (except on last chunk)
        v16bf b0n = b0c, b1n = b1c, b2n = b2c, b3n = b3c;
        if (kc + 1 < NKC) {
            const bf16_t* bn = bbase + (size_t)(kc + 1) * 65536;
            b0n = *(const v16bf*)(bn);
            b1n = *(const v16bf*)(bn + 512);
            b2n = *(const v16bf*)(bn + 1024);
            b3n = *(const v16bf*)(bn + 1536);
        }

        // A fragments for both row-tiles from LDS (ISA layout K sub-blocks)
        const int k0 = kc * 32 + hi * 8;
        const bf16_t* pa0 = &As[(l)      * AROW + k0];
        const bf16_t* pa1 = &As[(16 + l) * AROW + k0];
        v8bf a0lo = *(const v8bf*)(pa0);
        v8bf a0hi = *(const v8bf*)(pa0 + 16);
        v8bf a1lo = *(const v8bf*)(pa1);
        v8bf a1hi = *(const v8bf*)(pa1 + 16);
        v16bf a0 = __builtin_shufflevector(a0lo, a0hi,
                     0, 1, 2, 3, 4, 5, 6, 7, 8, 9, 10, 11, 12, 13, 14, 15);
        v16bf a1 = __builtin_shufflevector(a1lo, a1hi,
                     0, 1, 2, 3, 4, 5, 6, 7, 8, 9, 10, 11, 12, 13, 14, 15);

        acc_i0 = __builtin_amdgcn_wmma_f32_16x16x32_bf16(false, a0, false, b0c, (short)0, acc_i0, false, false);
        acc_i1 = __builtin_amdgcn_wmma_f32_16x16x32_bf16(false, a1, false, b0c, (short)0, acc_i1, false, false);
        acc_f0 = __builtin_amdgcn_wmma_f32_16x16x32_bf16(false, a0, false, b1c, (short)0, acc_f0, false, false);
        acc_f1 = __builtin_amdgcn_wmma_f32_16x16x32_bf16(false, a1, false, b1c, (short)0, acc_f1, false, false);
        acc_g0 = __builtin_amdgcn_wmma_f32_16x16x32_bf16(false, a0, false, b2c, (short)0, acc_g0, false, false);
        acc_g1 = __builtin_amdgcn_wmma_f32_16x16x32_bf16(false, a1, false, b2c, (short)0, acc_g1, false, false);
        acc_o0 = __builtin_amdgcn_wmma_f32_16x16x32_bf16(false, a0, false, b3c, (short)0, acc_o0, false, false);
        acc_o1 = __builtin_amdgcn_wmma_f32_16x16x32_bf16(false, a1, false, b3c, (short)0, acc_o1, false, false);

        b0c = b0n; b1c = b1n; b2c = b2n; b3c = b3n;
    }

    // ---- gates + state update for both row-tiles
    const int jn = j0 + l;
    const float bi = bias[jn];
    const float bf = bias[jn + H_];
    const float bg = bias[jn + 2 * H_];
    const float bo = bias[jn + 3 * H_];

    gate_epilogue(acc_i0, acc_f0, acc_g0, acc_o0, row0,      hi, jn,
                  bi, bf, bg, bo, resets_t, c_state, ys_t, h_next);
    gate_epilogue(acc_i1, acc_f1, acc_g1, acc_o1, row0 + 16, hi, jn,
                  bi, bf, bg, bo, resets_t, c_state, ys_t, h_next);
}

// ---------------- host-side launch ----------------
extern "C" void kernel_launch(void* const* d_in, const int* in_sizes, int n_in,
                              void* d_out, int out_size, void* d_ws, size_t ws_size,
                              hipStream_t stream) {
    const float* ins    = (const float*)d_in[0];   // T x B x H
    const int*   resets = (const int*)  d_in[1];   // T x B
    const float* c0     = (const float*)d_in[2];   // B x H
    const float* h0     = (const float*)d_in[3];   // B x H
    const float* Wx     = (const float*)d_in[4];   // H x 4H
    const float* Wh     = (const float*)d_in[5];   // H x 4H
    const float* bias   = (const float*)d_in[6];   // 4H

    float* out = (float*)d_out;
    float* ys  = out;                                   // T*B*H
    float* cT  = out + (size_t)T_ * B_ * H_;            // B*H
    float* hT  = cT + (size_t)B_ * H_;                  // B*H

    // workspace carve-up
    char* ws = (char*)d_ws;
    bf16_t* ins_bf   = (bf16_t*)ws; ws += (size_t)T_ * B_ * H_ * sizeof(bf16_t);   // 64 MB
    bf16_t* Wfrag    = (bf16_t*)ws; ws += (size_t)K2 * G4 * sizeof(bf16_t);        // 4 MB
    bf16_t* hbuf0    = (bf16_t*)ws; ws += (size_t)B_ * H_ * sizeof(bf16_t);        // 256 KB
    bf16_t* hbuf1    = (bf16_t*)ws; ws += (size_t)B_ * H_ * sizeof(bf16_t);        // 256 KB
    float*  c_state  = (float*)ws;  ws += (size_t)B_ * H_ * sizeof(float);         // 512 KB
    bf16_t* zero_row = (bf16_t*)ws; ws += (size_t)H_ * sizeof(bf16_t);             // 1 KB

    // one-time conversions / packing / state init
    cvt_f32_to_bf16 <<<8192, 256, 0, stream>>>(ins, ins_bf, T_ * B_ * H_);
    pack_weights_frag<<<2048, 256, 0, stream>>>(Wx, Wh, Wfrag);
    init_state      <<<512,  256, 0, stream>>>(c0, h0, c_state, hbuf0, zero_row);

    const size_t lds_bytes = (size_t)MROWS * AROW * sizeof(bf16_t);  // 66,048 B

    // sequential scan: 256 dependent step kernels (stream-ordered)
    for (int t = 0; t < T_; ++t) {
        bf16_t* hprev = (t & 1) ? hbuf1 : hbuf0;
        bf16_t* hnext = (t & 1) ? hbuf0 : hbuf1;
        lstm_step<<<64, 128, lds_bytes, stream>>>(
            ins_bf + (size_t)t * B_ * H_,
            resets + (size_t)t * B_,
            Wfrag, bias,
            hprev, zero_row, hnext, c_state,
            ys + (size_t)t * B_ * H_);
    }

    finalize_out<<<512, 256, 0, stream>>>(c_state, ys + (size_t)(T_ - 1) * B_ * H_, cT, hT);
}
